// WaveRNN_12000138625472
// MI455X (gfx1250) — matrix operand order repaired and checked by reading, not compiled, not run on hardware
//
#include <hip/hip_runtime.h>
#include <hip/hip_bf16.h>
#include <stdint.h>

#define B_   8
#define T_   5120
#define TMEL 24
#define MEL_ 80
#define CD_  128
#define RNN_ 512
#define G3_  1536
#define KCAT 544   /* 512 + 32 aux */
#define M_ROWS (B_ * T_)   /* 40960 */

typedef _Float16 v16h __attribute__((ext_vector_type(16)));
typedef float    v8f  __attribute__((ext_vector_type(8)));
typedef unsigned int u32x4 __attribute__((ext_vector_type(4)));
typedef int          i32x4 __attribute__((ext_vector_type(4)));
typedef int          i32x8 __attribute__((ext_vector_type(8)));

#if __has_builtin(__builtin_amdgcn_tensor_load_to_lds) && __has_builtin(__builtin_amdgcn_s_wait_tensorcnt)
#define HAVE_TDM 1
#else
#define HAVE_TDM 0
#endif

// ---------------------------------------------------------------------------
// WMMA fragment helpers (wave32; A/B k order is contiguous k = 16*(lane/16) + e)
// ---------------------------------------------------------------------------
__device__ __forceinline__ v16h load_a_frag(const _Float16* base, int ld) {
  // A 16x32 f16 row-major: M = lane%16 ; lane<16 -> K {0..15} ; lane>=16 -> K {16..31}
  int lane = threadIdx.x & 31;
  int m  = lane & 15;
  int hi = lane >> 4;
  const _Float16* row = base + (long)m * ld + hi * 16;
  v16h a;
#pragma unroll
  for (int e = 0; e < 8; ++e) a[e] = row[e];
#pragma unroll
  for (int e = 0; e < 8; ++e) a[8 + e] = row[8 + e];
  return a;
}

// B fragment from swizzled storage: 32 contiguous bytes per lane -> 2x ds_load_b128
__device__ __forceinline__ v16h load_b_frag_sw(const _Float16* base) {
  int lane = threadIdx.x & 31;
  const _Float16* p = base + lane * 16;
  v16h b;
#pragma unroll
  for (int e = 0; e < 16; ++e) b[e] = p[e];
  return b;
}

__device__ __forceinline__ void store_d_tile(float* dst, int ld, v8f d) {
  // D 16x16 f32: VGPR r -> M = (lane>=16 ? 8 : 0) + r, N = lane%16
  int lane = threadIdx.x & 31;
  int n  = lane & 15;
  int mb = (lane >> 4) * 8;
#pragma unroll
  for (int r = 0; r < 8; ++r) dst[(long)(mb + r) * ld + n] = d[r];
}

__device__ __forceinline__ v8f wmma32(v16h a, v16h b, v8f c) {
  return __builtin_amdgcn_wmma_f32_16x16x32_f16(false, a, false, b, (short)0, c, false, false);
}

// ---------------------------------------------------------------------------
// TDM tile load: global f16 -> LDS (ISA §8 descriptor layout)
// ---------------------------------------------------------------------------
#if HAVE_TDM
__device__ __forceinline__ void tdm_load(unsigned lds_off, const _Float16* gptr,
                                         unsigned tile_w, unsigned tile_h,
                                         unsigned row_stride) {
  unsigned long long ga = (unsigned long long)(uintptr_t)gptr;
  u32x4 g0;
  g0[0] = 1u;                                    // count=1, user mode, no gather
  g0[1] = lds_off;                               // lds_addr
  g0[2] = (unsigned)(ga & 0xffffffffu);          // global_addr[31:0]
  g0[3] = (unsigned)((ga >> 32) & 0x01ffffffu) | (2u << 30);  // addr[56:32] | type=2
  unsigned dim0 = (row_stride > tile_w) ? row_stride : tile_w;
  unsigned dim1 = 0x00100000u;                   // generous (never OOB here)
  i32x8 g1;
  g1[0] = (int)(1u << 16);                       // workgroup_mask=0, data_size=1 (2B)
  g1[1] = (int)((dim0 & 0xffffu) << 16);         // tensor_dim0[15:0]
  g1[2] = (int)(((dim0 >> 16) & 0xffffu) | ((dim1 & 0xffffu) << 16));
  g1[3] = (int)(((dim1 >> 16) & 0xffffu) | ((tile_w & 0xffffu) << 16));
  g1[4] = (int)(tile_h & 0xffffu);               // tile_dim1 ; tile_dim2=0
  g1[5] = (int)row_stride;                       // tensor_dim0_stride[31:0]
  g1[6] = 0;
  g1[7] = 0;
  i32x4 z4 = {0, 0, 0, 0};
#if __clang_major__ >= 23
  i32x8 z8 = {0, 0, 0, 0, 0, 0, 0, 0};
  __builtin_amdgcn_tensor_load_to_lds(g0, g1, z4, z4, z8, 0);
#else
  __builtin_amdgcn_tensor_load_to_lds(g0, g1, z4, z4, 0);
#endif
}
#endif

__device__ __forceinline__ unsigned lds_offset(const void* p) {
  return (unsigned)(uintptr_t)p;   // low 32 bits of flat LDS addr == LDS offset
}

// ---------------------------------------------------------------------------
// WMMA GEMM: C(MxN) f32 = A(MxK) f16 @ Bsw(KxN swizzled) f16 (+bias, opt relu)
// block = 256 thr (8 waves). Block tile 128x128; wave w: rows [16w,16w+16).
// Bsw layout: [K/32][N/128] tiles of 4096 f16 = [nt(8)][lane(32)][e(16)],
//   value = B[kc*32 + (lane>>4)*16 + e][nb*128 + nt*16 + (lane&15)]
// ---------------------------------------------------------------------------
__global__ void k_gemm(const _Float16* __restrict__ A, const _Float16* __restrict__ Bsw,
                       const float* __restrict__ bias, float* __restrict__ C,
                       int M, int N, int K, int relu) {
  __shared__ __attribute__((aligned(16))) _Float16 sA[2][128 * 32];
  __shared__ __attribute__((aligned(16))) _Float16 sB[2][32 * 128];
  const int bn = blockIdx.x, bm = blockIdx.y;
  const int row0 = bm * 128, col0 = bn * 128;
  const int tid = threadIdx.x, wave = tid >> 5, lane = tid & 31;
  const int nk = K >> 5, nb128 = N >> 7;
  v8f acc[8] = {};

#if HAVE_TDM
  if (tid < 32) {
    tdm_load(lds_offset(&sA[0][0]), A + (long)row0 * K, 32, 128, (unsigned)K);
    tdm_load(lds_offset(&sB[0][0]), Bsw + (long)bn * 4096, 4096, 1, 4096);
  }
  for (int kc = 0; kc < nk; ++kc) {
    const int buf = kc & 1;
    if (tid < 32) {
      if (kc + 1 < nk) {
        tdm_load(lds_offset(&sA[buf ^ 1][0]),
                 A + (long)row0 * K + (kc + 1) * 32, 32, 128, (unsigned)K);
        tdm_load(lds_offset(&sB[buf ^ 1][0]),
                 Bsw + ((long)(kc + 1) * nb128 + bn) * 4096, 4096, 1, 4096);
        __builtin_amdgcn_s_wait_tensorcnt(2);   // oldest stage complete
      } else {
        __builtin_amdgcn_s_wait_tensorcnt(0);
      }
    }
    __syncthreads();
    v16h af = load_a_frag(&sA[buf][(wave * 16) * 32], 32);
#pragma unroll
    for (int nt = 0; nt < 8; ++nt) {
      v16h bf = load_b_frag_sw(&sB[buf][(nt * 32) * 16]);
      acc[nt] = wmma32(af, bf, acc[nt]);
    }
    __syncthreads();
  }
#else
  for (int kc = 0; kc < nk; ++kc) {
    for (int i = tid; i < 128 * 32; i += 256)
      sA[0][i] = A[(long)(row0 + (i >> 5)) * K + kc * 32 + (i & 31)];
    for (int i = tid; i < 32 * 128; i += 256)
      sB[0][i] = Bsw[((long)kc * nb128 + bn) * 4096 + i];
    __syncthreads();
    v16h af = load_a_frag(&sA[0][(wave * 16) * 32], 32);
#pragma unroll
    for (int nt = 0; nt < 8; ++nt) {
      v16h bf = load_b_frag_sw(&sB[0][(nt * 32) * 16]);
      acc[nt] = wmma32(af, bf, acc[nt]);
    }
    __syncthreads();
  }
#endif
  // epilogue
  const int n  = lane & 15;
  const int mb = (lane >> 4) * 8;
#pragma unroll
  for (int nt = 0; nt < 8; ++nt) {
    int col = col0 + nt * 16 + n;
    float bv = bias ? bias[col] : 0.f;
#pragma unroll
    for (int r = 0; r < 8; ++r) {
      long rr = row0 + wave * 16 + mb + r;
      float v = acc[nt][r] + bv;
      if (relu) v = fmaxf(v, 0.f);
      C[rr * N + col] = v;
    }
  }
}

// ---------------------------------------------------------------------------
// Persistent GRU scan. 32 blocks x 128 threads. Block w owns h-cols [16w,16w+16)
// and its z/r/n gate slices of Wh, held in LDS in fragment-contiguous layout
// sw[g][kc][lane][e] (48 KB), filled once from f32 weights.
// Per step: 3 waves x 16 WMMA (K=512); wave 3 prefetches next mx rows.
// h double-buffered in global; one grid sense-barrier per step.
// ---------------------------------------------------------------------------
__device__ __forceinline__ void gsync(unsigned* bar, unsigned nb) {
  __syncthreads();
  __threadfence();
  if (threadIdx.x == 0) {
    volatile unsigned* cnt = bar;
    volatile unsigned* gen = bar + 1;
    unsigned g = *gen;
    unsigned prev = atomicAdd((unsigned*)cnt, 1u);
    if (prev == nb - 1) {
      *cnt = 0;
      __threadfence();
      atomicAdd((unsigned*)gen, 1u);
    } else {
      while (*gen == g) {
#if defined(__HIP_DEVICE_COMPILE__)
        __builtin_amdgcn_s_sleep(2);
#endif
      }
    }
    __threadfence();   // acquire: don't serve stale WGP$ lines for h reads
  }
  __syncthreads();
}

__device__ __forceinline__ float sigmoidf_(float x) { return 1.f / (1.f + __expf(-x)); }

#define SCAN_NB 32
__global__ void k_gru_scan(const float* __restrict__ mx, const float* __restrict__ wh,
                           const float* __restrict__ b1, const float* __restrict__ x_in,
                           float* __restrict__ x_out, _Float16* __restrict__ h16,
                           float* __restrict__ h32, unsigned* bar) {
  __shared__ __attribute__((aligned(16))) _Float16 sw[3 * 16 * 32 * 16];  // 48 KB
  __shared__ float stile[3][16][16];                                      // 3 KB
  const int w = blockIdx.x;        // owns h cols [16w, 16w+16)
  const int tid = threadIdx.x, wave = tid >> 5, lane = tid & 31;

  // fragment-contiguous Wh slices: sw[((g*16+kc)*32+lane)*16+e]
  //   = Wh[kc*32 + (lane>>4)*16 + e][g*512 + 16w + (lane&15)]
  for (int i = tid; i < 3 * 16 * 32 * 16; i += 128) {
    int e = i & 15, ln = (i >> 4) & 31, kc = (i >> 9) & 15, g = i >> 13;
    int k = kc * 32 + ((ln >> 4) << 4) + e;
    int n = ln & 15;
    sw[i] = (_Float16)wh[(long)k * G3_ + g * RNN_ + 16 * w + n];
  }
  // zero h ping-pong buffers (h16 is 16x512 per buf: rows 8..15 stay zero = WMMA pad)
  for (int i = tid; i < 2 * 16 * 16; i += 128) {
    int buf = i >> 8, r = (i >> 4) & 15, c = i & 15;
    h16[(long)buf * 16 * RNN_ + r * RNN_ + 16 * w + c] = (_Float16)0.f;
  }
  for (int i = tid; i < 2 * 8 * 16; i += 128) {
    int buf = i >> 7, r = (i >> 4) & 7, c = i & 15;
    h32[(long)buf * 8 * RNN_ + r * RNN_ + 16 * w + c] = 0.f;
  }
  gsync(bar, SCAN_NB);

  for (int t = 0; t < T_; ++t) {
    const int rb = t & 1, wb = rb ^ 1;
    const _Float16* hsrc = h16 + (long)rb * 16 * RNN_;
    if (wave < 3) {
      v8f acc = {};
#pragma unroll 4
      for (int kc = 0; kc < 16; ++kc) {
        v16h a = load_a_frag(hsrc + kc * 32, RNN_);
        v16h b = load_b_frag_sw(&sw[((wave * 16 + kc) * 32) * 16]);
        acc = wmma32(a, b, acc);
      }
      store_d_tile(&stile[wave][0][0], 16, acc);
    } else if (t + 1 < T_) {
      // idle wave: prefetch next step's mx gate rows (global_prefetch_b8)
      int bb = lane & 7, g = lane >> 3;
      if (g < 3)
        __builtin_prefetch(&mx[((long)bb * T_ + t + 1) * G3_ + g * RNN_ + 16 * w], 0, 1);
    }
    __syncthreads();
    {   // 128 threads = (batch 0..7) x (col 0..15)
      int bb = tid >> 4, c = tid & 15, hc = 16 * w + c;
      long r = (long)bb * T_ + t;
      const float* mrow = mx + r * G3_;
      float mhz = stile[0][bb][c] + b1[hc];
      float mhr = stile[1][bb][c] + b1[RNN_ + hc];
      float mhn = stile[2][bb][c] + b1[2 * RNN_ + hc];
      float z  = sigmoidf_(mrow[hc] + mhz);
      float rg = sigmoidf_(mrow[RNN_ + hc] + mhr);
      float nn = tanhf(mrow[2 * RNN_ + hc] + rg * mhn);
      float hold = h32[(long)rb * 8 * RNN_ + bb * RNN_ + hc];
      float hn = z * hold + (1.f - z) * nn;
      h32[(long)wb * 8 * RNN_ + bb * RNN_ + hc] = hn;
      h16[(long)wb * 16 * RNN_ + bb * RNN_ + hc] = (_Float16)hn;
      x_out[r * RNN_ + hc] = x_in[r * RNN_ + hc] + hn;
    }
    gsync(bar, SCAN_NB);
  }
}

__global__ void k_init_bar(unsigned* bar) { bar[0] = 0u; bar[1] = 0u; }

// ---------------------------------------------------------------------------
// Small / frontend kernels (f32, negligible cost)
// ---------------------------------------------------------------------------
__global__ void k_upsmooth(const float* __restrict__ in, float* __restrict__ out,
                           const float* __restrict__ kv, int klen, int scale,
                           int Tin, long n) {
  const int Tout = Tin * scale;
  for (long idx = (long)blockIdx.x * blockDim.x + threadIdx.x; idx < n;
       idx += (long)gridDim.x * blockDim.x) {
    int c = (int)(idx % MEL_);
    long rt = idx / MEL_;
    int t = (int)(rt % Tout);
    int b = (int)(rt / Tout);
    float s = 0.f;
    for (int j = 0; j < klen; ++j) {
      int ti = t + j - klen / 2;
      if (ti >= 0 && ti < Tout)
        s += kv[j] * in[((long)b * Tin + ti / scale) * MEL_ + c];
    }
    out[idx] = s;
  }
}

__global__ void k_conv_in(const float* __restrict__ mels, const float* __restrict__ W,
                          float* __restrict__ out) {
  int idx = blockIdx.x * blockDim.x + threadIdx.x;   // 160*128
  if (idx >= 160 * CD_) return;
  int c = idx & 127, r = idx >> 7;
  int b = r / 20, t = r % 20;
  float s = 0.f;
  for (int w = 0; w < 5; ++w)
    for (int i = 0; i < MEL_; ++i)
      s += mels[((long)b * TMEL + t + w) * MEL_ + i] * W[(w * MEL_ + i) * CD_ + c];
  out[idx] = s;
}

__global__ void k_bn_prep(const float* __restrict__ x, const float* __restrict__ g,
                          const float* __restrict__ bb, float* __restrict__ scale,
                          float* __restrict__ shift) {
  int c = threadIdx.x;   // 128
  float s = 0.f, s2 = 0.f;
  for (int r = 0; r < 160; ++r) {
    float v = x[r * CD_ + c];
    s += v; s2 += v * v;
  }
  float m = s / 160.f, var = s2 / 160.f - m * m;
  float sc = g[c] * rsqrtf(var + 1e-3f);
  scale[c] = sc;
  shift[c] = bb[c] - m * sc;
}

__global__ void k_bn_apply(const float* __restrict__ x, const float* __restrict__ scale,
                           const float* __restrict__ shift, const float* __restrict__ res,
                           float* __restrict__ y, int relu) {
  int idx = blockIdx.x * blockDim.x + threadIdx.x;
  if (idx >= 160 * CD_) return;
  int c = idx & 127;
  float v = scale[c] * x[idx] + shift[c];
  if (res) v += res[idx];
  if (relu) v = fmaxf(v, 0.f);
  y[idx] = v;
}

__global__ void k_mm_small(const float* __restrict__ A, const float* __restrict__ W,
                           const float* __restrict__ bias, float* __restrict__ C) {
  int idx = blockIdx.x * blockDim.x + threadIdx.x;
  if (idx >= 160 * CD_) return;
  int c = idx & 127, r = idx >> 7;
  float s = bias ? bias[c] : 0.f;
  for (int k = 0; k < CD_; ++k) s += A[r * CD_ + k] * W[k * CD_ + c];
  C[idx] = s;
}

// converters / packers ------------------------------------------------------
__global__ void k_cvt16(const float* __restrict__ src, _Float16* __restrict__ dst, long n) {
  for (long i = (long)blockIdx.x * blockDim.x + threadIdx.x; i < n;
       i += (long)gridDim.x * blockDim.x)
    dst[i] = (_Float16)src[i];
}

// f32 (KxN row-major) weight -> swizzled f16 Bsw tiles; zero-fill rows >= Ksrc
__global__ void k_pack_bsw(const float* __restrict__ W, _Float16* __restrict__ dst,
                           int Ksrc, int K, int N) {
  long total = (long)K * N;
  int nb128 = N >> 7;
  for (long idx = (long)blockIdx.x * blockDim.x + threadIdx.x; idx < total;
       idx += (long)gridDim.x * blockDim.x) {
    int e  = (int)(idx & 15);
    int ln = (int)((idx >> 4) & 31);
    int nt = (int)((idx >> 9) & 7);
    long tile = idx >> 12;
    int nb = (int)(tile % nb128);
    int kc = (int)(tile / nb128);
    int k = kc * 32 + ((ln >> 4) << 4) + e;
    int n = nb * 128 + nt * 16 + (ln & 15);
    dst[idx] = (k < Ksrc) ? (_Float16)W[(long)k * N + n] : (_Float16)0.f;
  }
}

// P[r, 0..127] = [audio | mel_up(80) | aux0(32) | zero pad]
__global__ void k_pack_in(const float* __restrict__ audio, const float* __restrict__ m2,
                          const float* __restrict__ aux, _Float16* __restrict__ P) {
  for (long idx = (long)blockIdx.x * blockDim.x + threadIdx.x; idx < (long)M_ROWS * 128;
       idx += (long)gridDim.x * blockDim.x) {
    int col = (int)(idx & 127);
    long row = idx >> 7;
    int b = (int)(row / T_), t = (int)(row % T_);
    float v;
    if (col == 0)       v = audio[(long)b * T_ + t];
    else if (col < 81)  v = m2[((long)b * 6144 + 512 + t) * MEL_ + (col - 1)];
    else if (col < 113) v = aux[((long)b * 20 + (t >> 8)) * CD_ + (col - 81)];
    else                v = 0.f;
    P[idx] = (_Float16)v;
  }
}

// P[r, 0..543] = [x(512 f32) | aux chunk (32)]
__global__ void k_pack_xa(const float* __restrict__ x, const float* __restrict__ aux,
                          int aux_off, _Float16* __restrict__ P) {
  for (long idx = (long)blockIdx.x * blockDim.x + threadIdx.x; idx < (long)M_ROWS * KCAT;
       idx += (long)gridDim.x * blockDim.x) {
    int col = (int)(idx % KCAT);
    long row = idx / KCAT;
    float v;
    if (col < RNN_) {
      v = x[row * RNN_ + col];
    } else {
      int b = (int)(row / T_), t = (int)(row % T_);
      v = aux[((long)b * 20 + (t >> 8)) * CD_ + aux_off + (col - RNN_)];
    }
    P[idx] = (_Float16)v;
  }
}

// ---------------------------------------------------------------------------
// Orchestration
// ---------------------------------------------------------------------------
static inline size_t align256(size_t x) { return (x + 255) & ~(size_t)255; }

extern "C" void kernel_launch(void* const* d_in, const int* in_sizes, int n_in,
                              void* d_out, int out_size, void* d_ws, size_t ws_size,
                              hipStream_t stream) {
  (void)in_sizes; (void)n_in; (void)out_size; (void)ws_size;
  const float* audios     = (const float*)d_in[0];
  const float* mels       = (const float*)d_in[1];
  const float* conv_in_w  = (const float*)d_in[2];
  const float* bn_in_g    = (const float*)d_in[3];
  const float* bn_in_b    = (const float*)d_in[4];
  const float* rb_w1      = (const float*)d_in[5];
  const float* rb_g1      = (const float*)d_in[6];
  const float* rb_b1      = (const float*)d_in[7];
  const float* rb_w2      = (const float*)d_in[8];
  const float* rb_g2      = (const float*)d_in[9];
  const float* rb_b2      = (const float*)d_in[10];
  const float* conv_out_w = (const float*)d_in[11];
  const float* conv_out_b = (const float*)d_in[12];
  const float* mel_k0     = (const float*)d_in[13];
  const float* mel_k1     = (const float*)d_in[14];
  const float* mel_k2     = (const float*)d_in[15];
  const float* i_w        = (const float*)d_in[16];
  const float* i_b        = (const float*)d_in[17];
  const float* gru0_wx    = (const float*)d_in[18];
  const float* gru0_wh    = (const float*)d_in[19];
  const float* gru0_b     = (const float*)d_in[20];
  const float* gru1_wx    = (const float*)d_in[21];
  const float* gru1_wh    = (const float*)d_in[22];
  const float* gru1_b     = (const float*)d_in[23];
  const float* d0_w       = (const float*)d_in[24];
  const float* d0_b       = (const float*)d_in[25];
  const float* d1_w       = (const float*)d_in[26];
  const float* d1_b       = (const float*)d_in[27];
  const float* p_w        = (const float*)d_in[28];
  const float* p_b        = (const float*)d_in[29];
  float* out = (float*)d_out;

  // ---- workspace bump allocator ----
  char* base = (char*)d_ws; size_t off = 0;
  auto alloc = [&](size_t bytes) { void* p = base + off; off = align256(off + bytes); return p; };
  float*    MX   = (float*)alloc((size_t)M_ROWS * G3_ * 4);
  float*    XA   = (float*)alloc((size_t)M_ROWS * RNN_ * 4);
  float*    XB   = (float*)alloc((size_t)M_ROWS * RNN_ * 4);
  _Float16* P16  = (_Float16*)alloc((size_t)M_ROWS * KCAT * 2);
  _Float16* iw16 = (_Float16*)alloc((size_t)128 * RNN_ * 2);
  _Float16* g0wx = (_Float16*)alloc((size_t)RNN_ * G3_ * 2);
  _Float16* g1wx = (_Float16*)alloc((size_t)KCAT * G3_ * 2);
  _Float16* d0w  = (_Float16*)alloc((size_t)KCAT * RNN_ * 2);
  _Float16* d1w  = (_Float16*)alloc((size_t)KCAT * RNN_ * 2);
  _Float16* pw   = (_Float16*)alloc((size_t)RNN_ * RNN_ * 2);
  float*    M0   = (float*)alloc((size_t)B_ * 96 * MEL_ * 4);
  float*    M1   = (float*)alloc((size_t)B_ * 384 * MEL_ * 4);
  float*    M2   = (float*)alloc((size_t)B_ * 6144 * MEL_ * 4);
  float*    FA   = (float*)alloc((size_t)160 * CD_ * 4);
  float*    FH   = (float*)alloc((size_t)160 * CD_ * 4);
  float*    FH2  = (float*)alloc((size_t)160 * CD_ * 4);
  float*    AUX  = (float*)alloc((size_t)160 * CD_ * 4);
  float*    SC   = (float*)alloc((size_t)CD_ * 4);
  float*    SH   = (float*)alloc((size_t)CD_ * 4);
  _Float16* H16  = (_Float16*)alloc((size_t)2 * 16 * RNN_ * 2);
  float*    H32  = (float*)alloc((size_t)2 * 8 * RNN_ * 4);
  unsigned* BAR  = (unsigned*)alloc(256);

  const int BT = 256;
  auto blocks = [](long n, int bt) { return (unsigned)((n + bt - 1) / bt); };

  // ---- mel upsampling chain ----
  k_upsmooth<<<blocks((long)B_ * 96 * MEL_, BT), BT, 0, stream>>>(mels, M0, mel_k0, 9, 4, TMEL, (long)B_ * 96 * MEL_);
  k_upsmooth<<<blocks((long)B_ * 384 * MEL_, BT), BT, 0, stream>>>(M0, M1, mel_k1, 9, 4, 96, (long)B_ * 384 * MEL_);
  k_upsmooth<<<blocks((long)B_ * 6144 * MEL_, BT), BT, 0, stream>>>(M1, M2, mel_k2, 33, 16, 384, (long)B_ * 6144 * MEL_);

  // ---- conv frontend + res blocks ----
  k_conv_in<<<blocks(160 * CD_, BT), BT, 0, stream>>>(mels, conv_in_w, FH);
  k_bn_prep<<<1, CD_, 0, stream>>>(FH, bn_in_g, bn_in_b, SC, SH);
  k_bn_apply<<<blocks(160 * CD_, BT), BT, 0, stream>>>(FH, SC, SH, nullptr, FA, 1);
  for (int i = 0; i < 10; ++i) {
    k_mm_small<<<blocks(160 * CD_, BT), BT, 0, stream>>>(FA, rb_w1 + (long)i * CD_ * CD_, nullptr, FH);
    k_bn_prep<<<1, CD_, 0, stream>>>(FH, rb_g1 + i * CD_, rb_b1 + i * CD_, SC, SH);
    k_bn_apply<<<blocks(160 * CD_, BT), BT, 0, stream>>>(FH, SC, SH, nullptr, FH, 1);
    k_mm_small<<<blocks(160 * CD_, BT), BT, 0, stream>>>(FH, rb_w2 + (long)i * CD_ * CD_, nullptr, FH2);
    k_bn_prep<<<1, CD_, 0, stream>>>(FH2, rb_g2 + i * CD_, rb_b2 + i * CD_, SC, SH);
    k_bn_apply<<<blocks(160 * CD_, BT), BT, 0, stream>>>(FH2, SC, SH, FA, FA, 0);
  }
  k_mm_small<<<blocks(160 * CD_, BT), BT, 0, stream>>>(FA, conv_out_w, conv_out_b, AUX);

  // ---- weight swizzle+convert to f16 Bsw tiles ----
  k_pack_bsw<<<blocks((long)128 * RNN_, BT), BT, 0, stream>>>(i_w, iw16, 113, 128, RNN_);
  k_pack_bsw<<<blocks((long)RNN_ * G3_, BT), BT, 0, stream>>>(gru0_wx, g0wx, RNN_, RNN_, G3_);
  k_pack_bsw<<<blocks((long)KCAT * G3_, BT), BT, 0, stream>>>(gru1_wx, g1wx, KCAT, KCAT, G3_);
  k_pack_bsw<<<blocks((long)KCAT * RNN_, BT), BT, 0, stream>>>(d0_w, d0w, KCAT, KCAT, RNN_);
  k_pack_bsw<<<blocks((long)KCAT * RNN_, BT), BT, 0, stream>>>(d1_w, d1w, KCAT, KCAT, RNN_);
  k_pack_bsw<<<blocks((long)RNN_ * RNN_, BT), BT, 0, stream>>>(p_w, pw, RNN_, RNN_, RNN_);

  // ---- x0 = [audio|mel|aux0] @ i_w + i_b ----
  k_pack_in<<<blocks((long)M_ROWS * 128, BT), BT, 0, stream>>>(audios, M2, AUX, P16);
  k_gemm<<<dim3(RNN_ / 128, M_ROWS / 128), 256, 0, stream>>>(P16, iw16, i_b, XA, M_ROWS, RNN_, 128, 0);

  // ---- GRU0: mx0 = x0 @ wx + b0 ; scan ----
  k_cvt16<<<blocks((long)M_ROWS * RNN_, BT), BT, 0, stream>>>(XA, P16, (long)M_ROWS * RNN_);
  k_gemm<<<dim3(G3_ / 128, M_ROWS / 128), 256, 0, stream>>>(P16, g0wx, gru0_b, MX, M_ROWS, G3_, RNN_, 0);
  k_init_bar<<<1, 1, 0, stream>>>(BAR);
  k_gru_scan<<<SCAN_NB, 128, 0, stream>>>(MX, gru0_wh, gru0_b + G3_, XA, XB, H16, H32, BAR);

  // ---- GRU1: mx1 = [x1|aux1] @ wx + b0 ; scan ----
  k_pack_xa<<<blocks((long)M_ROWS * KCAT, BT), BT, 0, stream>>>(XB, AUX, 32, P16);
  k_gemm<<<dim3(G3_ / 128, M_ROWS / 128), 256, 0, stream>>>(P16, g1wx, gru1_b, MX, M_ROWS, G3_, KCAT, 0);
  k_init_bar<<<1, 1, 0, stream>>>(BAR);
  k_gru_scan<<<SCAN_NB, 128, 0, stream>>>(MX, gru1_wh, gru1_b + G3_, XB, XA, H16, H32, BAR);

  // ---- dense head ----
  k_pack_xa<<<blocks((long)M_ROWS * KCAT, BT), BT, 0, stream>>>(XA, AUX, 64, P16);
  k_gemm<<<dim3(RNN_ / 128, M_ROWS / 128), 256, 0, stream>>>(P16, d0w, d0_b, XB, M_ROWS, RNN_, KCAT, 1);
  k_pack_xa<<<blocks((long)M_ROWS * KCAT, BT), BT, 0, stream>>>(XB, AUX, 96, P16);
  k_gemm<<<dim3(RNN_ / 128, M_ROWS / 128), 256, 0, stream>>>(P16, d1w, d1_b, XA, M_ROWS, RNN_, KCAT, 1);
  k_cvt16<<<blocks((long)M_ROWS * RNN_, BT), BT, 0, stream>>>(XA, P16, (long)M_ROWS * RNN_);
  k_gemm<<<dim3(RNN_ / 128, M_ROWS / 128), 256, 0, stream>>>(P16, pw, p_b, out, M_ROWS, RNN_, RNN_, 0);
}
